// StructLNN_83399674954175
// MI455X (gfx1250) — matrix operand order repaired and checked by reading, not compiled, not run on hardware
//
#include <hip/hip_runtime.h>

typedef __bf16 bf16_t;
typedef __bf16 bf16x8  __attribute__((ext_vector_type(8)));
typedef __bf16 bf16x16 __attribute__((ext_vector_type(16)));
typedef float  v8f     __attribute__((ext_vector_type(8)));

#define B_SZ  256
#define S_SZ  4096
#define INDIM 43
#define HID   64
#define NCLS  5

__device__ __forceinline__ bf16_t f2bf(float f) {
  unsigned int u = __builtin_bit_cast(unsigned int, f);
  u = (u + 0x7fffu + ((u >> 16) & 1u)) >> 16;
  unsigned short s = (unsigned short)u;
  return __builtin_bit_cast(bf16_t, s);
}

// Branchless transcendentals on the raw TRANS units (v_tanh_f32 / v_exp_f32 /
// v_rcp_f32) -- no OCML slow paths, no divergent EXEC save/restore.
__device__ __forceinline__ float fast_tanh(float x) {
#if __has_builtin(__builtin_amdgcn_tanhf)
  return __builtin_amdgcn_tanhf(x);        // CDNA5 V_TANH_F32 (single TRANS op)
#else
  float e = __builtin_amdgcn_exp2f(2.885390082f * x);   // exp(2x)
  return 1.0f - 2.0f * __builtin_amdgcn_rcpf(e + 1.0f);
#endif
}
__device__ __forceinline__ float fast_sigmoid(float x) {
#if __has_builtin(__builtin_amdgcn_tanhf)
  // sigmoid(x) = 0.5*tanh(x/2) + 0.5 : one TRANS instead of exp+rcp
  return __builtin_fmaf(0.5f, __builtin_amdgcn_tanhf(0.5f * x), 0.5f);
#else
  return __builtin_amdgcn_rcpf(1.0f + __builtin_amdgcn_exp2f(-1.442695041f * x));
#endif
}

// -------------------------------------------------------------------------
// Stage 1: LN(x) -> @W_proj+b -> LN -> SiLU, written bf16 to feat[S][B][64]
// -------------------------------------------------------------------------
__global__ __launch_bounds__(256) void feat_kernel(
    const float* __restrict__ x, const float* __restrict__ ln_in_g,
    const float* __restrict__ ln_in_b, const float* __restrict__ W_proj,
    const float* __restrict__ b_proj, const float* __restrict__ ln_p_g,
    const float* __restrict__ ln_p_b, bf16_t* __restrict__ featbuf) {
  __shared__ float Wp[INDIM * HID];
  __shared__ float lng[INDIM], lnb[INDIM];
  __shared__ float xr[4][INDIM + 1];
  __shared__ float ob[4][HID];

  const int tid = threadIdx.x;
  for (int t = tid; t < INDIM * HID; t += 256) Wp[t] = W_proj[t];
  if (tid < INDIM) { lng[tid] = ln_in_g[tid]; lnb[tid] = ln_in_b[tid]; }

  const int g = tid >> 6;   // row within block (4 rows/block)
  const int c = tid & 63;   // output feature column
  const long r = (long)blockIdx.x * 4 + g;   // row in [0, B*S)
  if (c < INDIM) xr[g][c] = x[r * INDIM + c];
  __syncthreads();

  // LayerNorm over input dim (population variance), redundantly per thread
  float mu = 0.f;
  #pragma unroll
  for (int k = 0; k < INDIM; ++k) mu += xr[g][k];
  mu *= (1.0f / INDIM);
  float var = 0.f;
  #pragma unroll
  for (int k = 0; k < INDIM; ++k) { float d = xr[g][k] - mu; var += d * d; }
  var *= (1.0f / INDIM);
  const float rs = rsqrtf(var + 1e-5f);

  float acc = b_proj[c];
  #pragma unroll
  for (int k = 0; k < INDIM; ++k) {
    float xn = (xr[g][k] - mu) * rs * lng[k] + lnb[k];
    acc += xn * Wp[k * HID + c];
  }
  ob[g][c] = acc;
  __syncthreads();

  // LayerNorm over the 64 projected features, then SiLU
  float mu2 = 0.f;
  #pragma unroll
  for (int k = 0; k < HID; ++k) mu2 += ob[g][k];
  mu2 *= (1.0f / HID);
  float v2 = 0.f;
  #pragma unroll
  for (int k = 0; k < HID; ++k) { float d = ob[g][k] - mu2; v2 += d * d; }
  v2 *= (1.0f / HID);
  float y = (acc - mu2) * rsqrtf(v2 + 1e-5f) * ln_p_g[c] + ln_p_b[c];
  float sv = y * fast_sigmoid(y);   // SiLU

  const long s = r & (S_SZ - 1);
  const long b = r >> 12;
  featbuf[(s * B_SZ + b) * HID + c] = f2bf(sv);
}

// -------------------------------------------------------------------------
// Stage 2: recurrent scan. One WG per 16 batch rows. 4 waves; wave w owns
// output columns [16w,16w+16) of ALL four gates: 16 bf16 WMMAs per step.
// Weights live in VGPRs (16 B-fragments), z=[feat_t|h] lives in LDS.
// h_t is written back over feat_t's global slot (becomes hs for the head).
// -------------------------------------------------------------------------
__global__ __launch_bounds__(128) void scan_kernel(
    const float* __restrict__ dt,
    const float* __restrict__ W_ff1, const float* __restrict__ b_ff1,
    const float* __restrict__ W_ff2, const float* __restrict__ b_ff2,
    const float* __restrict__ W_ta,  const float* __restrict__ b_ta,
    const float* __restrict__ W_tb,  const float* __restrict__ b_tb,
    bf16_t* __restrict__ featbuf) {
  constexpr int ZSTR = 136;  // bf16 elems per z row (272B: bank-staggered, 16B-aligned)
  __shared__ __align__(16) bf16_t z_lds[16 * ZSTR];
  __shared__ float dt_lds[16];

  const int tid  = threadIdx.x;
  const int lane = tid & 31;
  const int w    = tid >> 5;            // wave id == N tile 0..3
  const int b0   = blockIdx.x * 16;     // batch slab base

  const int ln15  = lane & 15;
  const int kbase = lane & 16;          // lane half * 16 (0 / 16); halved into koff
  const int koff  = kbase >> 1;         // A-frag K offset (0 / 8)
  const int colh  = w * 16 + ln15;      // hidden column 0..63
  const int crow  = (lane >> 4) << 3;   // C-layout row offset (0 / 8)

  // ---- preload all 16 weight B-fragments (bf16) into registers ----
  const float* Wg[4] = {W_ff1, W_ff2, W_ta, W_tb};
  bf16x16 Bf[4][4];
  #pragma unroll
  for (int g = 0; g < 4; ++g) {
    #pragma unroll
    for (int kb = 0; kb < 4; ++kb) {
      bf16x16 f;
      #pragma unroll
      for (int i = 0; i < 16; ++i)
        f[i] = f2bf(Wg[g][(kb * 32 + koff + (i & 7) + ((i >> 3) << 4)) * HID + colh]);
      Bf[g][kb] = f;
    }
  }
  const float bias0 = b_ff1[colh];
  const float bias1 = b_ff2[colh];
  const float bias2 = b_ta[colh];
  const float bias3 = b_tb[colh];

  // ---- init LDS: h=0, feat step 0, dt step 0 ----
  const int row8 = tid >> 3, chunk = tid & 7;
  {
    #pragma unroll
    for (int i = 0; i < 8; ++i)
      z_lds[row8 * ZSTR + 64 + chunk * 8 + i] = f2bf(0.0f);
    *(uint4*)&z_lds[row8 * ZSTR + chunk * 8] =
        *(const uint4*)(featbuf + ((size_t)(b0 + row8)) * HID + chunk * 8);
    if (tid < 16) dt_lds[tid] = dt[(size_t)(b0 + tid) * S_SZ];
  }

  float hnew[8];

  for (int s = 0; s < S_SZ; ++s) {
    __syncthreads();  // z_lds / dt_lds complete for step s

    float dtt[8];
    #pragma unroll
    for (int j = 0; j < 8; ++j) dtt[j] = dt_lds[j + crow] * 10.0f;

    // A fragments from LDS (ISA 16-bit A layout: K {0..7,16..23} / {8..15,24..31})
    bf16x16 Af[4];
    #pragma unroll
    for (int kb = 0; kb < 4; ++kb) {
      bf16x8 lo = *(const bf16x8*)&z_lds[ln15 * ZSTR + kb * 32 + koff];
      bf16x8 hi = *(const bf16x8*)&z_lds[ln15 * ZSTR + kb * 32 + koff + 16];
      Af[kb] = __builtin_shufflevector(lo, hi, 0,1,2,3,4,5,6,7,8,9,10,11,12,13,14,15);
    }

    // prefetch next step's feat/dt into registers while WMMAs run
    uint4 nf = {0u, 0u, 0u, 0u};
    float ndt = 0.0f;
    if (s + 1 < S_SZ) {
      nf = *(const uint4*)(featbuf + ((size_t)(s + 1) * B_SZ + b0 + row8) * HID + chunk * 8);
      if (tid < 16) ndt = dt[(size_t)(b0 + tid) * S_SZ + (s + 1)];
      if (s + 2 < S_SZ)
        __builtin_prefetch(featbuf + ((size_t)(s + 2) * B_SZ + b0 + row8) * HID + chunk * 8, 0, 1);
    }

    // 4 gates x 4 K-blocks = 16 x v_wmma_f32_16x16x32_bf16
    v8f acc0, acc1, acc2, acc3;
    #pragma unroll
    for (int j = 0; j < 8; ++j) { acc0[j] = bias0; acc1[j] = bias1; acc2[j] = bias2; acc3[j] = bias3; }
    #pragma unroll
    for (int kb = 0; kb < 4; ++kb) {
      acc0 = __builtin_amdgcn_wmma_f32_16x16x32_bf16(false, Af[kb], false, Bf[0][kb], (short)0, acc0, false, false);
      acc1 = __builtin_amdgcn_wmma_f32_16x16x32_bf16(false, Af[kb], false, Bf[1][kb], (short)0, acc1, false, false);
      acc2 = __builtin_amdgcn_wmma_f32_16x16x32_bf16(false, Af[kb], false, Bf[2][kb], (short)0, acc2, false, false);
      acc3 = __builtin_amdgcn_wmma_f32_16x16x32_bf16(false, Af[kb], false, Bf[3][kb], (short)0, acc3, false, false);
    }

    // h = tanh(ff1)*(1-t) + t*tanh(ff2),  t = sigmoid(ta*10dt + tb)
    // (branchless: 3x v_tanh_f32 per element)
    #pragma unroll
    for (int j = 0; j < 8; ++j) {
      float f1 = fast_tanh(acc0[j]);
      float f2 = fast_tanh(acc1[j]);
      float ti = fast_sigmoid(__builtin_fmaf(acc2[j], dtt[j], acc3[j]));
      hnew[j] = __builtin_fmaf(ti, f2 - f1, f1);
    }

    __syncthreads();  // all reads of step-s z/dt done

    // critical path first: install h into z (columns 64..127) for next step
    #pragma unroll
    for (int j = 0; j < 8; ++j)
      z_lds[(j + crow) * ZSTR + 64 + colh] = f2bf(hnew[j]);
    if (s + 1 < S_SZ) {
      *(uint4*)&z_lds[row8 * ZSTR + chunk * 8] = nf;
      if (tid < 16) dt_lds[tid] = ndt;
    }

    // fire-and-forget: dump h tile to global (reuses feat[s] slot -> hs buffer)
    {
      bf16_t* hp = featbuf + ((size_t)s * B_SZ + b0) * HID;
      #pragma unroll
      for (int j = 0; j < 8; ++j)
        hp[(size_t)(j + crow) * HID + colh] = f2bf(hnew[j]);
    }
  }
}

// -------------------------------------------------------------------------
// Stage 3: logits = hs @ W_head + b_head   ([B*S,64] x [64,5])
// -------------------------------------------------------------------------
__global__ __launch_bounds__(256) void head_kernel(
    const bf16_t* __restrict__ hs, const float* __restrict__ W_head,
    const float* __restrict__ b_head, float* __restrict__ out) {
  __shared__ float Wh[HID * NCLS];
  __shared__ float bh[NCLS];
  const int tid = threadIdx.x;
  for (int t = tid; t < HID * NCLS; t += 256) Wh[t] = W_head[t];
  if (tid < NCLS) bh[tid] = b_head[tid];
  __syncthreads();

  const size_t t = (size_t)blockIdx.x * 256 + tid;  // t = s*256 + b (coalesced hs reads)
  const int  b = (int)(t & 255);
  const long s = (long)(t >> 8);

  float acc[NCLS];
  #pragma unroll
  for (int c = 0; c < NCLS; ++c) acc[c] = bh[c];

  const uint4* hv = (const uint4*)(hs + t * HID);
  #pragma unroll
  for (int q = 0; q < 8; ++q) {
    uint4 u = hv[q];
    unsigned int uu[4] = {u.x, u.y, u.z, u.w};
    #pragma unroll
    for (int d = 0; d < 4; ++d) {
      int k = (q * 4 + d) * 2;
      float lo = __builtin_bit_cast(float, uu[d] << 16);
      float hi = __builtin_bit_cast(float, uu[d] & 0xffff0000u);
      #pragma unroll
      for (int c = 0; c < NCLS; ++c)
        acc[c] += lo * Wh[k * NCLS + c] + hi * Wh[(k + 1) * NCLS + c];
    }
  }
  float* op = out + ((size_t)b * S_SZ + s) * NCLS;
  #pragma unroll
  for (int c = 0; c < NCLS; ++c) op[c] = acc[c];
}

// -------------------------------------------------------------------------
extern "C" void kernel_launch(void* const* d_in, const int* in_sizes, int n_in,
                              void* d_out, int out_size, void* d_ws, size_t ws_size,
                              hipStream_t stream) {
  const float* x       = (const float*)d_in[0];
  const float* dt      = (const float*)d_in[1];
  const float* ln_in_g = (const float*)d_in[2];
  const float* ln_in_b = (const float*)d_in[3];
  const float* W_proj  = (const float*)d_in[4];
  const float* b_proj  = (const float*)d_in[5];
  const float* ln_p_g  = (const float*)d_in[6];
  const float* ln_p_b  = (const float*)d_in[7];
  const float* W_ff1   = (const float*)d_in[8];
  const float* b_ff1   = (const float*)d_in[9];
  const float* W_ff2   = (const float*)d_in[10];
  const float* b_ff2   = (const float*)d_in[11];
  const float* W_ta    = (const float*)d_in[12];
  const float* b_ta    = (const float*)d_in[13];
  const float* W_tb    = (const float*)d_in[14];
  const float* b_tb    = (const float*)d_in[15];
  const float* W_head  = (const float*)d_in[16];
  const float* b_head  = (const float*)d_in[17];
  float* out = (float*)d_out;

  // feat/hs buffer: [S][B][HID] bf16 = 128 MiB in workspace
  bf16_t* featbuf = (bf16_t*)d_ws;

  feat_kernel<<<(B_SZ * S_SZ) / 4, 256, 0, stream>>>(
      x, ln_in_g, ln_in_b, W_proj, b_proj, ln_p_g, ln_p_b, featbuf);
  scan_kernel<<<B_SZ / 16, 128, 0, stream>>>(
      dt, W_ff1, b_ff1, W_ff2, b_ff2, W_ta, b_ta, W_tb, b_tb, featbuf);
  head_kernel<<<(B_SZ * S_SZ) / 256, 256, 0, stream>>>(
      featbuf, W_head, b_head, out);
}
